// U_sup_37263136260675
// MI455X (gfx1250) — matrix operand order repaired and checked by reading, not compiled, not run on hardware
//
#include <hip/hip_runtime.h>
#include <hip/hip_bf16.h>
#include <math.h>

// ---------------------------------------------------------------------------
// Problem constants (match reference)
// ---------------------------------------------------------------------------
static constexpr int G4   = 4;
static constexpr int NPER = 512;
static constexpr int DIMc = 256;
static constexpr int UNIVc= 512;
static constexpr int NCLSc= 64;
static constexpr int TOTN = 2048;   // G4*NPER

typedef __attribute__((ext_vector_type(16))) _Float16 v16h;
typedef __attribute__((ext_vector_type(8)))  float    v8f;
typedef unsigned int v4u __attribute__((ext_vector_type(4)));
typedef unsigned int v8u __attribute__((ext_vector_type(8)));

// ---------------------------------------------------------------------------
// WMMA fragment loaders per CDNA5 ISA 7.12.2 (wave32 layouts)
// 16-bit A matrix 16x32 (MxK):
//   lanes 0-15  (M=lane):    v0..3 -> K 0..7,   v4..7 -> K16..23
//   lanes 16-31 (M=lane-16): v0..3 -> K 8..15,  v4..7 -> K24..31
// 16-bit B matrix 32x16 (KxN):
//   lanes 0-15  (N=lane):    v0..3 -> K 0..7,   v4..7 -> K16..23
//   lanes 16-31 (N=lane-16): v0..3 -> K 8..15,  v4..7 -> K24..31
// f32 C/D 16x16: vgpr r: lanes0-15 -> (M=r,N=lane), lanes16-31 -> (M=r+8)
// ---------------------------------------------------------------------------
template <int TA>
__device__ inline v16h load_a_frag(const float* __restrict__ A, int lda, int m0, int k0) {
  int lane = threadIdx.x & 31;
  int m  = m0 + (lane & 15);
  int kb = (lane < 16) ? 0 : 8;
  v16h a;
#pragma unroll
  for (int v = 0; v < 8; ++v) {
    int k = k0 + ((v < 4) ? 0 : 16) + kb + 2 * (v & 3);
    if (TA) {   // A stored transposed: A[k][m], strided
      a[2 * v]     = (_Float16)A[(size_t)k * lda + m];
      a[2 * v + 1] = (_Float16)A[(size_t)(k + 1) * lda + m];
    } else {    // row-major: pair contiguous along k
      float2 p = *(const float2*)&A[(size_t)m * lda + k];
      a[2 * v]     = (_Float16)p.x;
      a[2 * v + 1] = (_Float16)p.y;
    }
  }
  return a;
}

template <int TB>
__device__ inline v16h load_b_frag(const float* __restrict__ B, int ldb, int k0, int n0) {
  int lane = threadIdx.x & 31;
  int n  = n0 + (lane & 15);
  int kb = (lane < 16) ? 0 : 8;
  v16h b;
#pragma unroll
  for (int v = 0; v < 8; ++v) {
    int k = k0 + ((v < 4) ? 0 : 16) + kb + 2 * (v & 3);
    if (TB) {   // logical B = stored^T: B[n][k], pair contiguous along k
      float2 p = *(const float2*)&B[(size_t)n * ldb + k];
      b[2 * v]     = (_Float16)p.x;
      b[2 * v + 1] = (_Float16)p.y;
    } else {    // row-major: strided along k
      b[2 * v]     = (_Float16)B[(size_t)k * ldb + n];
      b[2 * v + 1] = (_Float16)B[(size_t)(k + 1) * ldb + n];
    }
  }
  return b;
}

__device__ inline void store_c_frag(float* C, int ldc, int m0, int n0, const v8f& c) {
  int lane = threadIdx.x & 31;
  int n  = n0 + (lane & 15);
  int mb = (lane < 16) ? 0 : 8;
#pragma unroll
  for (int r = 0; r < 8; ++r)
    C[(size_t)(m0 + mb + r) * ldc + n] = c[r];
}

// ---------------------------------------------------------------------------
// fp32-in/fp32-out GEMM, f16 WMMA compute, f32 accumulate.
// One wave computes a 16x64 strip of C (4 independent accumulators -> 4
// v_wmma per K-step, no D->C RAW chain between consecutive WMMAs).
// blockDim = (32,4): 4 waves -> 4 consecutive M tiles. K % 32 == 0.
// ---------------------------------------------------------------------------
template <int TA, int TB>
__global__ void __launch_bounds__(128)
gemm_wmma(const float* __restrict__ A, const float* __restrict__ B,
          float* __restrict__ C, int M, int N, int K,
          int lda, int ldb, int ldc) {
  int tileM = blockIdx.y * 4 + threadIdx.y;   // wave-uniform
  if (tileM * 16 >= M) return;                // wave-uniform exit
  int m0 = tileM * 16;
  int n0 = blockIdx.x * 64;
  v8f acc0 = {}, acc1 = {}, acc2 = {}, acc3 = {};
  for (int k0 = 0; k0 < K; k0 += 32) {
    if (k0 + 32 < K) {  // next K-tile prefetch (global_prefetch_b8)
      __builtin_prefetch(TA ? &A[(size_t)(k0 + 32) * lda + m0]
                            : &A[(size_t)m0 * lda + (k0 + 32)], 0, 1);
      __builtin_prefetch(TB ? &B[(size_t)n0 * ldb + (k0 + 32)]
                            : &B[(size_t)(k0 + 32) * ldb + n0], 0, 1);
    }
    v16h a  = load_a_frag<TA>(A, lda, m0, k0);
    v16h b0 = load_b_frag<TB>(B, ldb, k0, n0);
    v16h b1 = load_b_frag<TB>(B, ldb, k0, n0 + 16);
    v16h b2 = load_b_frag<TB>(B, ldb, k0, n0 + 32);
    v16h b3 = load_b_frag<TB>(B, ldb, k0, n0 + 48);
    acc0 = __builtin_amdgcn_wmma_f32_16x16x32_f16(false, a, false, b0, (short)0, acc0, false, false);
    acc1 = __builtin_amdgcn_wmma_f32_16x16x32_f16(false, a, false, b1, (short)0, acc1, false, false);
    acc2 = __builtin_amdgcn_wmma_f32_16x16x32_f16(false, a, false, b2, (short)0, acc2, false, false);
    acc3 = __builtin_amdgcn_wmma_f32_16x16x32_f16(false, a, false, b3, (short)0, acc3, false, false);
  }
  store_c_frag(C, ldc, m0, n0,      acc0);
  store_c_frag(C, ldc, m0, n0 + 16, acc1);
  store_c_frag(C, ldc, m0, n0 + 32, acc2);
  store_c_frag(C, ldc, m0, n0 + 48, acc3);
}

// ---------------------------------------------------------------------------
// Block reduction helpers
// ---------------------------------------------------------------------------
__device__ inline float block_reduce_max(float v, float* red) {
  int t = threadIdx.x;
  red[t] = v; __syncthreads();
  for (int s = blockDim.x >> 1; s > 0; s >>= 1) {
    if (t < s) red[t] = fmaxf(red[t], red[t + s]);
    __syncthreads();
  }
  float r = red[0]; __syncthreads();
  return r;
}
__device__ inline float block_reduce_sum(float v, float* red) {
  int t = threadIdx.x;
  red[t] = v; __syncthreads();
  for (int s = blockDim.x >> 1; s > 0; s >>= 1) {
    if (t < s) red[t] += red[t + s];
    __syncthreads();
  }
  float r = red[0]; __syncthreads();
  return r;
}

// ---------------------------------------------------------------------------
// Elementwise / small kernels
// ---------------------------------------------------------------------------
__global__ void zero_f(float* p, int n) {
  for (int i = blockIdx.x * blockDim.x + threadIdx.x; i < n;
       i += gridDim.x * blockDim.x) p[i] = 0.0f;
}
__global__ void scale_f(float* x, float s, int n) {
  for (int i = blockIdx.x * blockDim.x + threadIdx.x; i < n;
       i += gridDim.x * blockDim.x) x[i] *= s;
}
__global__ void exp_f(const float* __restrict__ src, float* __restrict__ dst, int n) {
  for (int i = blockIdx.x * blockDim.x + threadIdx.x; i < n;
       i += gridDim.x * blockDim.x) dst[i] = expf(src[i]);
}

__global__ void softmax_rows(float* X, int ncols, float scale) {
  __shared__ float red[256];
  float* row = X + (size_t)blockIdx.x * ncols;
  int t = threadIdx.x;
  float m = -INFINITY;
  for (int j = t; j < ncols; j += 256) m = fmaxf(m, row[j] * scale);
  m = block_reduce_max(m, red);
  float s = 0.0f;
  for (int j = t; j < ncols; j += 256) {
    float e = expf(row[j] * scale - m);
    row[j] = e; s += e;
  }
  s = block_reduce_sum(s, red);
  float inv = 1.0f / s;
  for (int j = t; j < ncols; j += 256) row[j] *= inv;
}

__global__ void row_scale_dinv(const float* __restrict__ node_out,
                               float* __restrict__ attn, int dim, int ncols) {
  __shared__ float red[256];
  int row = blockIdx.x;
  const float* x = node_out + (size_t)row * dim;
  int t = threadIdx.x;
  float d = 0.0f;
  for (int j = t; j < dim; j += 256) d += x[j] * x[j];
  float dots  = block_reduce_sum(d, red);
  float norms = sqrtf(dots);
  float D     = 1.0f - dots / (norms * norms);
  float dinv  = 1.0f / (D + 1e-8f);
  float* a = attn + (size_t)row * ncols;
  for (int j = t; j < ncols; j += 256) a[j] *= dinv;
}

__global__ void lse_rows_sub(float* X, int ncols) {
  __shared__ float red[256];
  float* row = X + (size_t)blockIdx.x * ncols;
  int t = threadIdx.x;
  float m = -INFINITY;
  for (int j = t; j < ncols; j += 256) m = fmaxf(m, row[j]);
  m = block_reduce_max(m, red);
  float s = 0.0f;
  for (int j = t; j < ncols; j += 256) s += expf(row[j] - m);
  s = block_reduce_sum(s, red);
  float lse = m + logf(s);
  for (int j = t; j < ncols; j += 256) row[j] -= lse;
}
__global__ void lse_cols_sub(float* X, int nrows, int ncols) {
  __shared__ float red[256];
  int col = blockIdx.x;
  int t = threadIdx.x;
  float m = -INFINITY;
  for (int i = t; i < nrows; i += 256) m = fmaxf(m, X[(size_t)i * ncols + col]);
  m = block_reduce_max(m, red);
  float s = 0.0f;
  for (int i = t; i < nrows; i += 256) s += expf(X[(size_t)i * ncols + col] - m);
  s = block_reduce_sum(s, red);
  float lse = m + logf(s);
  for (int i = t; i < nrows; i += 256) X[(size_t)i * ncols + col] -= lse;
}

__global__ void prep_labels(const int* __restrict__ labels, int* labm, int* cnt) {
  int i = blockIdx.x * blockDim.x + threadIdx.x;
  if (i < TOTN) {
    int l = labels[i] - 1;            // values in [1,64] -> [0,63]
    labm[i] = l;
    atomicAdd(&cnt[(i >> 9) * NCLSc + l], 1);
  }
}

__global__ void build_M(const float* __restrict__ edgeD,
                        const int* __restrict__ labm, float* M64) {
  __shared__ float acc[NCLSc * NCLSc];   // 16 KB
  int g  = blockIdx.x >> 6;
  int rc = blockIdx.x & 63;
  for (int idx = threadIdx.x; idx < NCLSc * NCLSc; idx += blockDim.x) acc[idx] = 0.0f;
  __syncthreads();
  for (int r = 0; r < 8; ++r) {
    int i  = rc * 8 + r;
    int li = labm[g * NPER + i];
    const float* row = edgeD + ((size_t)g * NPER + i) * NPER;
    for (int j = threadIdx.x; j < NPER; j += blockDim.x) {
      int lj = labm[g * NPER + j];
      atomicAdd(&acc[li * NCLSc + lj], row[j]);
    }
  }
  __syncthreads();
  for (int idx = threadIdx.x; idx < NCLSc * NCLSc; idx += blockDim.x)
    atomicAdd(&M64[idx], acc[idx]);
}

__global__ void segsum_S(const float* __restrict__ U, const int* __restrict__ labm,
                         float* S) {
  int r0 = blockIdx.x * 8;
  for (int r = 0; r < 8; ++r) {
    int i = r0 + r;
    int l = labm[i];
    const float* u = U + (size_t)i * UNIVc;
    for (int j = threadIdx.x; j < UNIVc; j += 256)
      atomicAdd(&S[l * UNIVc + j], u[j]);
  }
}

__global__ void S_from_state(const float* __restrict__ lsg,
                             const int* __restrict__ cnt, float* S) {
  int a = blockIdx.x;
  float c0 = (float)cnt[0 * NCLSc + a];
  float c1 = (float)cnt[1 * NCLSc + a];
  float c2 = (float)cnt[2 * NCLSc + a];
  float c3 = (float)cnt[3 * NCLSc + a];
  for (int j = threadIdx.x; j < UNIVc; j += 256) {
    float s = 0.0f;
    s += c0 * expf(lsg[((size_t)0 * NCLSc + a) * UNIVc + j]);
    s += c1 * expf(lsg[((size_t)1 * NCLSc + a) * UNIVc + j]);
    s += c2 * expf(lsg[((size_t)2 * NCLSc + a) * UNIVc + j]);
    s += c3 * expf(lsg[((size_t)3 * NCLSc + a) * UNIVc + j]);
    S[a * UNIVc + j] = s;
  }
}

// ---------------------------------------------------------------------------
// Fused per-graph HiPPI sinkhorn. State 64x512 fp32 lives in LDS (128 KB
// dynamic). The TK tile is DMA'd into LDS by the Tensor Data Mover:
// D# descriptor per ISA ch.8 (count=1, type=2(image), data_size=4B,
// tile 32768x1, tensor_dim0=stride=32768), issued by wave 0, completion via
// s_wait_tensorcnt. 20 iterations of row-lse / count-weighted col-lse fused.
// ---------------------------------------------------------------------------
__global__ void __launch_bounds__(512)
hippi_sinkhorn(const float* __restrict__ TK, const int* __restrict__ cnt,
               float* __restrict__ lsg) {
  extern __shared__ float smem[];
  float* ls   = smem;                 // 64*512 floats at LDS offset 0
  float* logc = smem + NCLSc * UNIVc; // 64
  int g = blockIdx.x;
  int t = threadIdx.x;
  if (t < NCLSc) {
    int c = cnt[g * NCLSc + t];
    logc[t] = (c > 0) ? logf((float)c) : -1e30f;
  }
  if (t < 32) {  // wave 0 issues the TDM transfer: global TK -> LDS[0..128KB)
    unsigned long long ga = (unsigned long long)(const void*)TK;
    v4u g0; v8u g1; v4u g2; v4u g3;
    g0[0] = 1u;                                   // count=1 (valid descriptor)
    g0[1] = 0u;                                   // lds_addr = 0 (dynamic base)
    g0[2] = (unsigned)(ga & 0xFFFFFFFFu);         // global_addr[31:0]
    g0[3] = (unsigned)((ga >> 32) & 0x01FFFFFFu)  // global_addr[56:32]
          | 0x80000000u;                          // type=2 (bits 127:126)
    g1[0] = 0x20000u;        // workgroup_mask=0, data_size=2 (4B), no flags
    g1[1] = 0x80000000u;     // tensor_dim0[15:0]=0x8000 at bits 63:48
    g1[2] = 0x00010000u;     // tensor_dim0[31:16]=0, tensor_dim1=1
    g1[3] = 0x80000000u;     // tile_dim0=0x8000 (32768 elems) at bits 127:112
    g1[4] = 0x00000001u;     // tile_dim1=1, tile_dim2=0
    g1[5] = 0x00008000u;     // tensor_dim0_stride = 32768
    g1[6] = 0x80000000u;     // stride hi=0; tensor_dim1_stride[15:0]=0x8000
    g1[7] = 0u;              // tensor_dim1_stride[47:16]=0
    g2[0] = 1u; g2[1] = 0u; g2[2] = 0u; g2[3] = 0u;   // tensor_dim2=1, rest 0
    g3[0] = 0u; g3[1] = 0u; g3[2] = 0u; g3[3] = 0u;
    asm volatile("tensor_load_to_lds %0, %1, %2, %3"
                 :: "s"(g0), "s"(g1), "s"(g2), "s"(g3) : "memory");
    __builtin_amdgcn_s_wait_tensorcnt(0);
  }
  __syncthreads();
  for (int idx = t; idx < NCLSc * UNIVc; idx += 512)
    ls[idx] *= 200.0f;                 // /tau, tau = 1/200
  __syncthreads();
  int lane = t & 31, wid = t >> 5;
  for (int it = 0; it < 20; ++it) {
    // row normalize (distinct label rows; expanded rows are identical)
    for (int a = wid; a < NCLSc; a += 16) {
      float* r = ls + a * UNIVc;
      float m = -INFINITY;
      for (int j = lane; j < UNIVc; j += 32) m = fmaxf(m, r[j]);
      for (int off = 16; off; off >>= 1) m = fmaxf(m, __shfl_xor(m, off, 32));
      float s = 0.0f;
      for (int j = lane; j < UNIVc; j += 32) s += expf(r[j] - m);
      for (int off = 16; off; off >>= 1) s += __shfl_xor(s, off, 32);
      float lse = m + logf(s);
      for (int j = lane; j < UNIVc; j += 32) r[j] -= lse;
    }
    __syncthreads();
    // column normalize, weighted by per-graph label multiplicity
    {
      int j = t;
      float m = -INFINITY;
      for (int a = 0; a < NCLSc; ++a) m = fmaxf(m, ls[a * UNIVc + j] + logc[a]);
      float s = 0.0f;
      for (int a = 0; a < NCLSc; ++a) s += expf(ls[a * UNIVc + j] + logc[a] - m);
      float lse = m + logf(s);
      for (int a = 0; a < NCLSc; ++a) ls[a * UNIVc + j] -= lse;
    }
    __syncthreads();
  }
  for (int idx = t; idx < NCLSc * UNIVc; idx += 512)
    lsg[(size_t)g * NCLSc * UNIVc + idx] = ls[idx];
}

// Loss reductions
__global__ void loss_accum(const float* __restrict__ U0, const float* __restrict__ lsg,
                           const int* __restrict__ labm, float* acc) {
  __shared__ float red[256];
  int i = blockIdx.x;
  int g = i >> 9;
  int l = labm[i];
  const float* u  = U0  + (size_t)i * UNIVc;
  const float* lr = lsg + ((size_t)g * NCLSc + l) * UNIVc;
  int t = threadIdx.x;
  float s = 0.0f;
  for (int j = t; j < UNIVc; j += 256) {
    float d = u[j] - expf(lr[j]);
    s += d * d;
  }
  s = block_reduce_sum(s, red);
  if (t == 0) atomicAdd(&acc[0], s);
}
__global__ void sqnorm_accum(const float* __restrict__ P, int n, float* acc) {
  __shared__ float red[256];
  float s = 0.0f;
  for (int i = blockIdx.x * blockDim.x + threadIdx.x; i < n;
       i += gridDim.x * blockDim.x) s += P[i] * P[i];
  s = block_reduce_sum(s, red);
  if (threadIdx.x == 0) atomicAdd(&acc[1], s);
}
__global__ void finalize_loss(const float* acc, float* out) {
  out[0] = 0.1f * (acc[0] / (float)((size_t)TOTN * UNIVc)) + 1e-4f * sqrtf(acc[1]);
}

// ---------------------------------------------------------------------------
// Host orchestration
// ---------------------------------------------------------------------------
static inline void launch_gemm(const float* A, const float* B, float* C,
                               int M, int N, int K, int lda, int ldb, int ldc,
                               int tA, int tB, hipStream_t stream) {
  dim3 blk(32, 4);
  dim3 grd(N / 64, (M / 16 + 3) / 4);
  if (!tA && !tB)
    gemm_wmma<0,0><<<grd, blk, 0, stream>>>(A, B, C, M, N, K, lda, ldb, ldc);
  else if (!tA && tB)
    gemm_wmma<0,1><<<grd, blk, 0, stream>>>(A, B, C, M, N, K, lda, ldb, ldc);
  else
    gemm_wmma<1,0><<<grd, blk, 0, stream>>>(A, B, C, M, N, K, lda, ldb, ldc);
}

extern "C" void kernel_launch(void* const* d_in, const int* in_sizes, int n_in,
                              void* d_out, int out_size, void* d_ws, size_t ws_size,
                              hipStream_t stream) {
  const float* nodes   = (const float*)d_in[0];  // [4,512,256]
  const int*   labels  = (const int*)  d_in[1];  // [4,512]
  const float* U_param = (const float*)d_in[2];  // [512,256]
  const float* Wq      = (const float*)d_in[3];  // [256,256]
  const float* Wk      = (const float*)d_in[4];
  const float* Wv      = (const float*)d_in[5];
  float* out = (float*)d_out;

  float* wsf = (float*)d_ws;
  float* Qf   = wsf;                         // 2048*256
  float* Kf   = Qf   + (size_t)TOTN * DIMc;
  float* Vf   = Kf   + (size_t)TOTN * DIMc;
  float* attn = Vf   + (size_t)TOTN * DIMc;  // 2048*512 (attn -> edge_D -> U0)
  float* nout = attn + (size_t)TOTN * NPER;  // 2048*256
  float* Nls  = nout + (size_t)TOTN * DIMc;  // 2048*512
  float* Kmat = Nls  + (size_t)TOTN * UNIVc; // 512*512
  float* S    = Kmat + (size_t)UNIVc * UNIVc;// 64*512
  float* T    = S    + (size_t)NCLSc * UNIVc;
  float* TK   = T    + (size_t)NCLSc * UNIVc;
  float* lsg  = TK   + (size_t)NCLSc * UNIVc;       // 4*64*512
  float* M64  = lsg  + (size_t)G4 * NCLSc * UNIVc;  // 64*64
  float* acc  = M64  + (size_t)NCLSc * NCLSc;       // 2
  int*   labm = (int*)(acc + 8);                    // 2048
  int*   cnt  = labm + TOTN;                        // 4*64
  float* U0   = attn;

  // 1) QKV projections
  launch_gemm(nodes, Wq, Qf, TOTN, DIMc, DIMc, DIMc, DIMc, DIMc, 0, 0, stream);
  launch_gemm(nodes, Wk, Kf, TOTN, DIMc, DIMc, DIMc, DIMc, DIMc, 0, 0, stream);
  launch_gemm(nodes, Wv, Vf, TOTN, DIMc, DIMc, DIMc, DIMc, DIMc, 0, 0, stream);

  // 2) per-graph logits = Q_g @ K_g^T ; softmax(1/sqrt(256))
  for (int g = 0; g < G4; ++g)
    launch_gemm(Qf + (size_t)g * NPER * DIMc, Kf + (size_t)g * NPER * DIMc,
                attn + (size_t)g * NPER * NPER, NPER, NPER, DIMc,
                DIMc, DIMc, NPER, 0, 1, stream);
  softmax_rows<<<TOTN, 256, 0, stream>>>(attn, NPER, 0.0625f);

  // 3) node_out_g = attn_g @ V_g
  for (int g = 0; g < G4; ++g)
    launch_gemm(attn + (size_t)g * NPER * NPER, Vf + (size_t)g * NPER * DIMc,
                nout + (size_t)g * NPER * DIMc, NPER, DIMc, NPER,
                NPER, DIMc, DIMc, 0, 0, stream);

  // 4) edge_D = attn * D_inv(row)
  row_scale_dinv<<<TOTN, 256, 0, stream>>>(nout, attn, DIMc, NPER);

  // 5) labels -> labm, per-graph counts
  zero_f<<<4, 256, 0, stream>>>((float*)cnt, G4 * NCLSc);
  prep_labels<<<(TOTN + 255) / 256, 256, 0, stream>>>(labels, labm, cnt);

  // 6) M = C^T A C (rank-64 collapse of W^T A W)
  zero_f<<<16, 256, 0, stream>>>(M64, NCLSc * NCLSc);
  build_M<<<G4 * 64, 256, 0, stream>>>(attn, labm, M64);

  // 7) N = node_out @ U_param^T ; U0 = log_sinkhorn(N, tau=0.05, 20)
  launch_gemm(nout, U_param, Nls, TOTN, UNIVc, DIMc, DIMc, DIMc, UNIVc, 0, 1, stream);
  scale_f<<<512, 256, 0, stream>>>(Nls, 20.0f, TOTN * UNIVc);
  for (int it = 0; it < 20; ++it) {
    lse_rows_sub<<<TOTN, 256, 0, stream>>>(Nls, UNIVc);
    lse_cols_sub<<<UNIVc, 256, 0, stream>>>(Nls, TOTN, UNIVc);
  }
  exp_f<<<512, 256, 0, stream>>>(Nls, U0, TOTN * UNIVc);

  // 8) HiPPI: 50 iterations in collapsed 64-label space
  size_t shmem = (size_t)(NCLSc * UNIVc + NCLSc) * sizeof(float);  // ~132 KB
  for (int it = 0; it < 50; ++it) {
    if (it == 0) {
      zero_f<<<64, 256, 0, stream>>>(S, NCLSc * UNIVc);
      segsum_S<<<TOTN / 8, 256, 0, stream>>>(U0, labm, S);
    } else {
      S_from_state<<<NCLSc, 256, 0, stream>>>(lsg, cnt, S);
    }
    launch_gemm(M64, S, T,   NCLSc, UNIVc, NCLSc, NCLSc, UNIVc, UNIVc, 0, 0, stream);
    launch_gemm(S, T, Kmat,  UNIVc, UNIVc, NCLSc, UNIVc, UNIVc, UNIVc, 1, 0, stream);
    launch_gemm(T, Kmat, TK, NCLSc, UNIVc, UNIVc, UNIVc, UNIVc, UNIVc, 0, 0, stream);
    hippi_sinkhorn<<<G4, 512, shmem, stream>>>(TK, cnt, lsg);
  }

  // 9) loss
  zero_f<<<1, 64, 0, stream>>>(acc, 2);
  loss_accum<<<TOTN, 256, 0, stream>>>(U0, lsg, labm, acc);
  sqnorm_accum<<<128, 256, 0, stream>>>(U_param, UNIVc * DIMc, acc);
  finalize_loss<<<1, 1, 0, stream>>>(acc, out);

  (void)in_sizes; (void)n_in; (void)out_size; (void)ws_size;
}
